// SRGNN_52055003627522
// MI455X (gfx1250) — matrix operand order repaired and checked by reading, not compiled, not run on hardware
//
#include <hip/hip_runtime.h>
#include <hip/hip_bf16.h>
#include <math.h>

#define DFEAT 128

typedef __attribute__((ext_vector_type(8)))  __bf16 v8bf;
typedef __attribute__((ext_vector_type(16))) __bf16 v16bf;
typedef __attribute__((ext_vector_type(8)))  float  v8f;

__device__ __forceinline__ float wave_sum32(float v) {
#pragma unroll
  for (int o = 16; o > 0; o >>= 1) v += __shfl_xor(v, o, 32);
  return v;
}
__device__ __forceinline__ float sigmoidf(float x) { return 1.0f / (1.0f + expf(-x)); }

__device__ __forceinline__ v16bf cat8(v8bf lo, v8bf hi) {
  return __builtin_shufflevector(lo, hi, 0,1,2,3,4,5,6,7,8,9,10,11,12,13,14,15);
}

// ---------------------------------------------------------------------------
// Generic GEMM: C[M x Nc] = scale * (A[M x K] @ B[Nc x K]^T) + bias[Nc]
// bf16 operands, f32 accumulate via v_wmma_f32_16x16x32_bf16.
// Block: 256 threads = 8 waves; tile 64 rows x 128 cols; 4 WMMA subtiles/wave.
//
// B tile (128 rows x K bf16) is staged into LDS ONCE per block with
// GLOBAL_LOAD_ASYNC_TO_LDS_B128 (ASYNCcnt), removing the 64x redundant
// global re-fetch of B across row-blocks/waves. LDS rows are padded to
// K+8 elements so B-fragment ds_load_b128 reads are bank-conflict free.
//
// Fragment layouts per CDNA5 ISA 7.12.2:
//   A (16x32 bf16): lane l: M = l%16; elems 0..7 -> K = (l/16)*8 + e,
//                   elems 8..15 -> K = 16 + (l/16)*8 + (e-8)
//   B (32x16 bf16): lane l: N = l%16; elems 0..7 -> K = (l/16)*8 + e, elems 8..15 -> K = 16 + (l/16)*8 + (e-8)
//   C (16x16 f32):  lane l, elem r: M = (l/16)*8 + r; N = l%16
// ---------------------------------------------------------------------------
__global__ void gemm_bf16_abT(const __bf16* __restrict__ A, int lda,
                              const __bf16* __restrict__ Bm, int ldb,
                              float* __restrict__ Cf, __bf16* __restrict__ Cb,
                              int ldc, const float* __restrict__ bias, float scale,
                              int M, int Nc, int K) {
  extern __shared__ __bf16 Bs[];          // [128][K+8]
  const int ldsb = K + 8;                 // padded LDS row stride (elements)
  const int tid = threadIdx.x;
  const int n0 = blockIdx.x * 128;

  // ---- stage B tile into LDS (async global->LDS, 16B per lane per op) ----
  {
    const int chunks_per_row = K >> 3;                 // 16B chunks per row
    const int total = 128 * chunks_per_row;
    const unsigned lds_base = __builtin_amdgcn_groupstaticsize();
    for (int i = tid; i < total; i += 256) {
      const int r  = i / chunks_per_row;
      const int ck = i - r * chunks_per_row;
      const int rrow = min(n0 + r, Nc - 1);
      const __bf16* gsrc = Bm + (size_t)rrow * ldb + ck * 8;
      const unsigned lds_off =
          lds_base + (unsigned)(((unsigned)r * ldsb + ck * 8) * sizeof(__bf16));
      asm volatile("global_load_async_to_lds_b128 %0, %1, off"
                   :: "v"(lds_off), "v"(gsrc) : "memory");
    }
    asm volatile("s_wait_asynccnt 0x0" ::: "memory");
  }
  __syncthreads();

  // ---- WMMA compute: A from global (tiny, L2-hot), B from LDS ----
  const int lane = tid & 31;
  const int wave = tid >> 5;
  const int lh   = lane >> 4;
  const int l16  = lane & 15;
  const int m0   = blockIdx.y * 64 + (wave & 3) * 16;
  const int nn0  = (wave >> 2) * 64;      // tile-local column base

  const int mA = min(m0 + l16, M - 1);
  const __bf16* arow = A + (size_t)mA * lda;
  const __bf16* bloc[4];
#pragma unroll
  for (int t = 0; t < 4; ++t)
    bloc[t] = Bs + (size_t)(nn0 + t * 16 + l16) * ldsb;

  v8f acc[4];
#pragma unroll
  for (int t = 0; t < 4; ++t) acc[t] = (v8f){0.f,0.f,0.f,0.f,0.f,0.f,0.f,0.f};

  for (int kk = 0; kk < K; kk += 32) {
    const __bf16* ap = arow + kk;
    v16bf af = cat8(*(const v8bf*)(ap + lh * 8), *(const v8bf*)(ap + 16 + lh * 8));
#pragma unroll
    for (int t = 0; t < 4; ++t) {
      const __bf16* bp = bloc[t] + kk + lh * 8;
      v16bf bf = cat8(*(const v8bf*)(bp), *(const v8bf*)(bp + 16));
      acc[t] = __builtin_amdgcn_wmma_f32_16x16x32_bf16(
          false, af, false, bf, (short)0, acc[t], false, false);
    }
  }

#pragma unroll
  for (int t = 0; t < 4; ++t) {
    const int n = n0 + nn0 + t * 16 + l16;
    if (n >= Nc) continue;
    const float bsv = bias ? bias[n] : 0.0f;
#pragma unroll
    for (int r = 0; r < 8; ++r) {
      const int m = m0 + lh * 8 + r;
      if (m >= M) continue;
      const float v = acc[t][r] * scale + bsv;
      if (Cf) Cf[(size_t)m * ldc + n] = v;
      if (Cb) Cb[(size_t)m * ldc + n] = (__bf16)v;
    }
  }
}

// ---------------------------------------------------------------------------
// Wave-per-row gather + L2-normalize. iid==nullptr -> identity gather.
// ---------------------------------------------------------------------------
__global__ void gather_l2norm(const float* __restrict__ src, const int* __restrict__ iid,
                              float* __restrict__ f32out, __bf16* __restrict__ bfout,
                              int nrows) {
  const int wave = (blockIdx.x * blockDim.x + threadIdx.x) >> 5;
  const int lane = threadIdx.x & 31;
  if (wave >= nrows) return;
  const int srow = iid ? iid[wave] : wave;
  const float4 v = *(const float4*)(src + (size_t)srow * DFEAT + lane * 4);
  float ss = v.x * v.x + v.y * v.y + v.z * v.z + v.w * v.w;
  ss = wave_sum32(ss);
  const float inv = 1.0f / fmaxf(sqrtf(ss), 1e-12f);
  const float4 o = {v.x * inv, v.y * inv, v.z * inv, v.w * inv};
  if (f32out) *(float4*)(f32out + (size_t)wave * DFEAT + lane * 4) = o;
  if (bfout) {
    __bf16* p = bfout + (size_t)wave * DFEAT + lane * 4;
    p[0] = (__bf16)o.x; p[1] = (__bf16)o.y; p[2] = (__bf16)o.z; p[3] = (__bf16)o.w;
  }
}

__global__ void zero_f32(float* __restrict__ p, size_t n) {
  size_t i = (size_t)blockIdx.x * blockDim.x + threadIdx.x;
  if (i < n) p[i] = 0.0f;
}

__global__ void cvt_f32_bf16(const float* __restrict__ in, __bf16* __restrict__ out, int n) {
  int i = blockIdx.x * blockDim.x + threadIdx.x;
  if (i < n) out[i] = (__bf16)in[i];
}

// One thread per (edge, d): weighted scatter-add for both edge directions.
__global__ void edge_scatter(const float* __restrict__ feat,
                             const int* __restrict__ src, const int* __restrict__ dst,
                             const float* __restrict__ w,
                             float* __restrict__ m1, float* __restrict__ w1s,
                             float* __restrict__ m2, float* __restrict__ w2s, int E) {
  int idx = blockIdx.x * blockDim.x + threadIdx.x;
  int e = idx >> 7, d = idx & 127;
  if (e >= E) return;
  const float we = w[e];
  const int s = src[e], t = dst[e];
  atomicAdd(&m1[(size_t)t * DFEAT + d], feat[(size_t)s * DFEAT + d] * we);
  atomicAdd(&m2[(size_t)s * DFEAT + d], feat[(size_t)t * DFEAT + d] * we);
  if (d == 0) { atomicAdd(&w1s[t], we); atomicAdd(&w2s[s], we); }
}

// agg = (wsum>0) ? msum/max(wsum,1e-12) : 0, emitted as bf16
__global__ void agg_div_bf16(const float* __restrict__ msum, const float* __restrict__ wsum,
                             __bf16* __restrict__ out, int n) {
  int idx = blockIdx.x * blockDim.x + threadIdx.x;
  if (idx >= n * DFEAT) return;
  int row = idx >> 7;
  float ws = wsum[row];
  float v = (ws > 0.0f) ? msum[idx] / fmaxf(ws, 1e-12f) : 0.0f;
  out[idx] = (__bf16)v;
}

// Wave-per-row GRU cell + L2 norm. gi/gh are [n x 384] (r,z,n gate layout).
__global__ void gru_l2norm(const float* __restrict__ gi, const float* __restrict__ gh,
                           const float* __restrict__ h0,
                           float* __restrict__ f32out, __bf16* __restrict__ bfout, int nrows) {
  const int wave = (blockIdx.x * blockDim.x + threadIdx.x) >> 5;
  const int lane = threadIdx.x & 31;
  if (wave >= nrows) return;
  const float* giR = gi + (size_t)wave * 384;
  const float* ghR = gh + (size_t)wave * 384;
  float out[4];
  float ss = 0.0f;
#pragma unroll
  for (int j = 0; j < 4; ++j) {
    int d = lane * 4 + j;
    float r = sigmoidf(giR[d] + ghR[d]);
    float z = sigmoidf(giR[128 + d] + ghR[128 + d]);
    float nn = tanhf(giR[256 + d] + r * ghR[256 + d]);
    float h = h0[(size_t)wave * DFEAT + d];
    out[j] = (1.0f - z) * nn + z * h;
    ss += out[j] * out[j];
  }
  ss = wave_sum32(ss);
  const float inv = 1.0f / fmaxf(sqrtf(ss), 1e-12f);
#pragma unroll
  for (int j = 0; j < 4; ++j) {
    int d = lane * 4 + j;
    float v = out[j] * inv;
    if (f32out) f32out[(size_t)wave * DFEAT + d] = v;
    if (bfout) bfout[(size_t)wave * DFEAT + d] = (__bf16)v;
  }
}

// e[i] = sum_d sigmoid(fu[i,d] + fv[seg[i],d]) * We[d]   (wave per node)
__global__ void attn_e(const float* __restrict__ fu, const float* __restrict__ fv,
                       const int* __restrict__ segids, const float* __restrict__ We,
                       float* __restrict__ e, int n) {
  const int wave = (blockIdx.x * blockDim.x + threadIdx.x) >> 5;
  const int lane = threadIdx.x & 31;
  if (wave >= n) return;
  const int g = segids[wave];
  float acc = 0.0f;
#pragma unroll
  for (int j = 0; j < 4; ++j) {
    int d = lane * 4 + j;
    float x = fu[(size_t)wave * DFEAT + d] + fv[(size_t)g * DFEAT + d];
    acc += sigmoidf(x) * We[d];
  }
  acc = wave_sum32(acc);
  if (lane == 0) e[wave] = acc;
}

// Per-segment softmax over S consecutive nodes (one thread per segment).
__global__ void seg_softmax(const float* __restrict__ e, float* __restrict__ alpha,
                            int nseg, int S) {
  int g = blockIdx.x * blockDim.x + threadIdx.x;
  if (g >= nseg) return;
  float m = -3.4e38f;
  for (int s = 0; s < S; ++s) m = fmaxf(m, e[g * S + s]);
  float sum = 0.0f;
  for (int s = 0; s < S; ++s) sum += expf(e[g * S + s] - m);
  for (int s = 0; s < S; ++s) alpha[g * S + s] = expf(e[g * S + s] - m) / sum;
}

// sr_cat[g, 128 + d] = sum_s feat[g*S+s, d] * alpha[g*S+s]   (block per segment)
__global__ void seg_weighted_sum(const float* __restrict__ feat, const float* __restrict__ alpha,
                                 __bf16* __restrict__ sr_cat, int S) {
  int g = blockIdx.x, d = threadIdx.x;
  float acc = 0.0f;
  for (int s = 0; s < S; ++s) {
    int node = g * S + s;
    acc += feat[(size_t)node * DFEAT + d] * alpha[node];
  }
  sr_cat[(size_t)g * 256 + 128 + d] = (__bf16)acc;
}

// sr_cat[g, d] = feat_bf16[last[g], d]
__global__ void gather_last(const __bf16* __restrict__ featb, const int* __restrict__ last,
                            __bf16* __restrict__ sr_cat) {
  int g = blockIdx.x, d = threadIdx.x;
  sr_cat[(size_t)g * 256 + d] = featb[(size_t)last[g] * DFEAT + d];
}

// In-place row-wise log-softmax with online max/sum (block per row; 512 threads).
__global__ void row_logsoftmax(float* __restrict__ out, int ncols) {
  const int row = blockIdx.x, tid = threadIdx.x;
  float* r = out + (size_t)row * ncols;
  float m = -3.4e38f, s = 0.0f;
  for (int c = tid; c < ncols; c += blockDim.x) {
    float x = r[c];
    float mn = fmaxf(m, x);
    s = s * expf(m - mn) + expf(x - mn);
    m = mn;
  }
#pragma unroll
  for (int o = 16; o > 0; o >>= 1) {
    float mo = __shfl_xor(m, o, 32), so = __shfl_xor(s, o, 32);
    float mn = fmaxf(m, mo);
    s = s * expf(m - mn) + so * expf(mo - mn);
    m = mn;
  }
  __shared__ float sm[16], ssum[16];
  const int wave = tid >> 5, lane = tid & 31;
  if (lane == 0) { sm[wave] = m; ssum[wave] = s; }
  __syncthreads();
  if (wave == 0) {
    m = (lane < 16) ? sm[lane] : -3.4e38f;
    s = (lane < 16) ? ssum[lane] : 0.0f;
#pragma unroll
    for (int o = 16; o > 0; o >>= 1) {
      float mo = __shfl_xor(m, o, 32), so = __shfl_xor(s, o, 32);
      float mn = fmaxf(m, mo);
      s = s * expf(m - mn) + so * expf(mo - mn);
      m = mn;
    }
    if (lane == 0) { sm[0] = m; ssum[0] = logf(s); }
  }
  __syncthreads();
  m = sm[0];
  const float ls = ssum[0];
  for (int c = tid; c < ncols; c += blockDim.x) r[c] = r[c] - m - ls;
}

// ---------------------------------------------------------------------------
extern "C" void kernel_launch(void* const* d_in, const int* in_sizes, int n_in,
                              void* d_out, int out_size, void* d_ws, size_t ws_size,
                              hipStream_t stream) {
  (void)n_in; (void)out_size; (void)ws_size;
  const int N = in_sizes[0];            // 10240 nodes
  const int E = in_sizes[1];            // 9216 edges
  const int B = in_sizes[4];            // 1024 sessions
  const int ITEMS = in_sizes[6] / DFEAT;// 50000
  const int S = N / B;                  // 10

  const int*   iid     = (const int*)d_in[0];
  const int*   esrc    = (const int*)d_in[1];
  const int*   edst    = (const int*)d_in[2];
  const float* ew      = (const float*)d_in[3];
  const int*   last    = (const int*)d_in[4];
  const int*   segids  = (const int*)d_in[5];
  const float* emb     = (const float*)d_in[6];
  const float* W1      = (const float*)d_in[7];
  const float* W2      = (const float*)d_in[8];
  const float* w_ih    = (const float*)d_in[9];
  const float* w_hh    = (const float*)d_in[10];
  const float* b_ih    = (const float*)d_in[11];
  const float* b_hh    = (const float*)d_in[12];
  const float* Wu      = (const float*)d_in[13];
  const float* Wv      = (const float*)d_in[14];
  const float* bv      = (const float*)d_in[15];
  const float* We      = (const float*)d_in[16];
  const float* W_sr    = (const float*)d_in[17];
  float* out = (float*)d_out;

  // ---- workspace carve-up ----
  char* w = (char*)d_ws;
  size_t off = 0;
  auto alloc = [&](size_t bytes) -> char* {
    char* p = w + off;
    off = (off + bytes + 255) & ~(size_t)255;
    return p;
  };
  float*  feat_f   = (float*)alloc((size_t)N * DFEAT * 4);
  __bf16* feat_b   = (__bf16*)alloc((size_t)N * DFEAT * 2);
  float*  msum1    = (float*)alloc((size_t)N * DFEAT * 4); // msum1,msum2,wsum1,wsum2 contiguous
  float*  msum2    = (float*)alloc((size_t)N * DFEAT * 4);
  float*  wsum1    = (float*)alloc((size_t)N * 4);
  float*  wsum2    = (float*)alloc((size_t)N * 4);
  __bf16* agg1b    = (__bf16*)alloc((size_t)N * DFEAT * 2);
  __bf16* agg2b    = (__bf16*)alloc((size_t)N * DFEAT * 2);
  __bf16* hnb      = (__bf16*)alloc((size_t)N * 256 * 2);
  float*  gi       = (float*)alloc((size_t)N * 384 * 4);
  float*  gh       = (float*)alloc((size_t)N * 384 * 4);
  float*  feat2_f  = (float*)alloc((size_t)N * DFEAT * 4);
  __bf16* feat2_b  = (__bf16*)alloc((size_t)N * DFEAT * 2);
  float*  feat_u   = (float*)alloc((size_t)N * DFEAT * 4);
  float*  fv       = (float*)alloc((size_t)B * DFEAT * 4);
  float*  evec     = (float*)alloc((size_t)N * 4);
  float*  alpha    = (float*)alloc((size_t)N * 4);
  __bf16* sr_cat   = (__bf16*)alloc((size_t)B * 256 * 2);
  float*  sr_f     = (float*)alloc((size_t)B * DFEAT * 4);
  __bf16* sr_b     = (__bf16*)alloc((size_t)B * DFEAT * 2);
  __bf16* target_b = (__bf16*)alloc((size_t)ITEMS * DFEAT * 2);
  __bf16* W1b      = (__bf16*)alloc(128 * 128 * 2);
  __bf16* W2b      = (__bf16*)alloc(128 * 128 * 2);
  __bf16* wihb     = (__bf16*)alloc(384 * 256 * 2);
  __bf16* whhb     = (__bf16*)alloc(384 * 128 * 2);
  __bf16* Wub      = (__bf16*)alloc(128 * 128 * 2);
  __bf16* Wvb      = (__bf16*)alloc(128 * 128 * 2);
  __bf16* Wsrb     = (__bf16*)alloc(128 * 256 * 2);

  auto gemm = [&](const __bf16* A, int lda, const __bf16* Bm, int ldb,
                  float* Cf, __bf16* Cb, int ldc, const float* bias, float scale,
                  int M, int Nc, int K) {
    dim3 grid((Nc + 127) / 128, (M + 63) / 64);
    size_t shbytes = (size_t)128 * (K + 8) * sizeof(__bf16);
    gemm_bf16_abT<<<grid, 256, shbytes, stream>>>(A, lda, Bm, ldb, Cf, Cb, ldc,
                                                  bias, scale, M, Nc, K);
  };
  auto cvt = [&](const float* in, __bf16* o, int n) {
    cvt_f32_bf16<<<(n + 255) / 256, 256, 0, stream>>>(in, o, n);
  };

  // 0) zero atomic-accumulation region (msum1,msum2,wsum1,wsum2 are contiguous)
  {
    size_t zn = (size_t)2 * N * DFEAT + 2 * N;
    zero_f32<<<(unsigned)((zn + 255) / 256), 256, 0, stream>>>(msum1, zn);
  }
  // 1) feat = l2norm(embedding[iid]); target = l2norm(embedding) (bf16)
  gather_l2norm<<<(N + 7) / 8, 256, 0, stream>>>(emb, iid, feat_f, feat_b, N);
  gather_l2norm<<<(ITEMS + 7) / 8, 256, 0, stream>>>(emb, nullptr, nullptr, target_b, ITEMS);
  // 2) weights -> bf16
  cvt(W1, W1b, 128 * 128); cvt(W2, W2b, 128 * 128);
  cvt(w_ih, wihb, 384 * 256); cvt(w_hh, whhb, 384 * 128);
  cvt(Wu, Wub, 128 * 128); cvt(Wv, Wvb, 128 * 128);
  cvt(W_sr, Wsrb, 128 * 256);
  // 3) edge aggregation (weighted mean, both directions)
  edge_scatter<<<((size_t)E * DFEAT + 255) / 256, 256, 0, stream>>>(
      feat_f, esrc, edst, ew, msum1, wsum1, msum2, wsum2, E);
  agg_div_bf16<<<((size_t)N * DFEAT + 255) / 256, 256, 0, stream>>>(msum1, wsum1, agg1b, N);
  agg_div_bf16<<<((size_t)N * DFEAT + 255) / 256, 256, 0, stream>>>(msum2, wsum2, agg2b, N);
  // 4) neigh1 = agg1 @ W1.T -> hn[:, :128]; neigh2 = agg2 @ W2.T -> hn[:, 128:]
  gemm(agg1b, 128, W1b, 128, nullptr, hnb, 256, nullptr, 1.0f, N, 128, 128);
  gemm(agg2b, 128, W2b, 128, nullptr, hnb + 128, 256, nullptr, 1.0f, N, 128, 128);
  // 5) GRU: gi = hn @ w_ih.T + b_ih ; gh = feat @ w_hh.T + b_hh
  gemm(hnb, 256, wihb, 256, gi, nullptr, 384, b_ih, 1.0f, N, 384, 256);
  gemm(feat_b, 128, whhb, 128, gh, nullptr, 384, b_hh, 1.0f, N, 384, 128);
  gru_l2norm<<<(N + 7) / 8, 256, 0, stream>>>(gi, gh, feat_f, feat2_f, feat2_b, N);
  // 6) attention
  gemm(feat2_b, 128, Wub, 128, feat_u, nullptr, 128, nullptr, 1.0f, N, 128, 128);
  gather_last<<<B, 128, 0, stream>>>(feat2_b, last, sr_cat);
  gemm(sr_cat, 256, Wvb, 128, fv, nullptr, 128, bv, 1.0f, B, 128, 128);
  attn_e<<<(N + 7) / 8, 256, 0, stream>>>(feat_u, fv, segids, We, evec, N);
  seg_softmax<<<(B + 255) / 256, 256, 0, stream>>>(evec, alpha, B, S);
  seg_weighted_sum<<<B, 128, 0, stream>>>(feat2_f, alpha, sr_cat, S);
  // 7) sr = l2norm(concat(sr_l, sr_g) @ W_sr.T)
  gemm(sr_cat, 256, Wsrb, 256, sr_f, nullptr, 128, nullptr, 1.0f, B, 128, 256);
  gather_l2norm<<<(B + 7) / 8, 256, 0, stream>>>(sr_f, nullptr, nullptr, sr_b, B);
  // 8) logits = 12 * (sr @ target.T), then in-place row log-softmax
  gemm(sr_b, 128, target_b, 128, out, nullptr, ITEMS, nullptr, 12.0f, B, ITEMS, 128);
  row_logsoftmax<<<B, 512, 0, stream>>>(out, ITEMS);
}
